// MultiHeadAttention_63934883168651
// MI455X (gfx1250) — matrix operand order repaired and checked by reading, not compiled, not run on hardware
//
#include <hip/hip_runtime.h>
#include <hip/hip_bf16.h>

// ---------------- problem constants (from reference) ----------------
#define B_   4
#define LQ_  2048
#define LK_  2048
#define D_   512
#define H_   8
#define DH_  64
#define LDP  2052   // padded LDS row stride in floats (2048 + 4; multiple of 4)

typedef __attribute__((ext_vector_type(16))) _Float16 v16h;
typedef __attribute__((ext_vector_type(4)))  _Float16 v4h;
typedef __attribute__((ext_vector_type(8)))  float    v8f;

// ---------------- WMMA helper (CDNA5 16x16x32 f16 -> f32) ----------------
static __device__ __forceinline__ v8f wmma16(v16h a, v16h b, v8f c) {
  // 8 args: (neg_a, A, neg_b, B, c_mod, C, reuse_a, reuse_b)
  return __builtin_amdgcn_wmma_f32_16x16x32_f16(false, a, false, b, (short)0, c,
                                                false, false);
}

// A-matrix 16x32 (MxK), row-major f16 source (global or LDS). Layout:
// lane<16 holds K 0..15 for row M=lane; lane>=16 holds K 16..31 (contiguous).
static __device__ __forceinline__ v16h load_fragA_f16(const _Float16* p, int ld, int lane) {
  const int r  = lane & 15;
  const int hi = lane >> 4;
  const _Float16* row = p + (size_t)r * ld;
  v16h o;
#pragma unroll
  for (int i = 0; i < 16; ++i) o[i] = row[hi * 16 + i];
  return o;
}

// B-matrix 32x16 (KxN) when the source is stored N x K row-major (i.e. B^T
// contiguous: Wq/Wc rows, K-matrix rows, transposed-V rows). Lane holds column
// n=lane&15; lanes 0-15 hold K=0..15, lanes 16-31 hold K=16..31 (contiguous).
static __device__ __forceinline__ v16h load_fragB_nk(const _Float16* p, int ld, int lane) {
  const int n  = lane & 15;
  const int hi = lane >> 4;
  const _Float16* row = p + (size_t)n * ld + hi * 16;
  v16h o;
#pragma unroll
  for (int i = 0; i < 16; ++i) o[i] = row[i];
  return o;
}

// C/D 16x16 f32: VGPR r, lane<16 -> (M=r, N=lane); lane>=16 -> (M=r+8, N=lane-16).
static __device__ __forceinline__ void store_C_f32(float* p, int ld, int lane, v8f c) {
  const int n  = lane & 15;
  const int mb = (lane >> 4) * 8;
#pragma unroll
  for (int r = 0; r < 8; ++r) p[(size_t)(mb + r) * ld + n] = c[r];
}
static __device__ __forceinline__ void store_C_f16(_Float16* p, int ld, int lane, v8f c) {
  const int n  = lane & 15;
  const int mb = (lane >> 4) * 8;
#pragma unroll
  for (int r = 0; r < 8; ++r) p[(size_t)(mb + r) * ld + n] = (_Float16)c[r];
}

// ---------------- wave32 reductions ----------------
static __device__ __forceinline__ float wave_max32(float v) {
#pragma unroll
  for (int off = 16; off > 0; off >>= 1) v = fmaxf(v, __shfl_xor(v, off, 32));
  return v;
}
static __device__ __forceinline__ float wave_sum32(float v) {
#pragma unroll
  for (int off = 16; off > 0; off >>= 1) v += __shfl_xor(v, off, 32);
  return v;
}

// ---------------- kernel 1: f32 -> f16 convert ----------------
__global__ void mha_cvt_f16_kernel(const float* __restrict__ src,
                                   _Float16* __restrict__ dst, int n) {
  int i = blockIdx.x * blockDim.x + threadIdx.x;
  if (i < n) dst[i] = (_Float16)src[i];
}

// -------- kernel 1b: fused f32->f16 convert + transpose of V --------
// in : value (B,LK,D) f32     out: VT (B,H,DH,LK) f16
__global__ void mha_cvt_transpose_v_kernel(const float* __restrict__ v,
                                           _Float16* __restrict__ vt) {
  const size_t i = (size_t)blockIdx.x * blockDim.x + threadIdx.x;  // over B*H*DH*LK
  const int k    = (int)(i & (LK_ - 1));
  const size_t r = i >> 11;                 // / LK_
  const int d    = (int)(r % DH_);
  const int h    = (int)((r / DH_) % H_);
  const int b    = (int)(r / (DH_ * H_));
  vt[i] = (_Float16)v[((size_t)b * LK_ + k) * D_ + h * DH_ + d];
}

// ---------------- kernel 2/4: GEMM  C[m,n] = scale * sum_k A[m,k]*Bt[n,k] ----
// A: MxK f16 row-major.  Bt: NxK f16 row-major (computes A @ B^T).
// One wave per 16x64 output strip (4 accumulators reuse each A fragment 4x).
template <bool OUT_HALF>
__global__ void mha_gemm_nt_kernel(const _Float16* __restrict__ A,
                                   const _Float16* __restrict__ Bt,
                                   void* __restrict__ Cv,
                                   int M, int N, int K, float scale) {
  const int lane = threadIdx.x & 31;
  const int w    = threadIdx.x >> 5;
  const int m0 = blockIdx.x * 16;
  const int n0 = (blockIdx.y * 8 + w) * 64;
  if (m0 >= M || n0 >= N) return;   // wave-uniform: EXEC stays all-ones for WMMA

  const _Float16* arow = A + (size_t)m0 * K;
  v8f acc[4] = {{}, {}, {}, {}};
  for (int kk = 0; kk < K; kk += 32) {
    __builtin_prefetch(arow + kk + 64, 0, 3);   // WGP-scope prefetch, 2 tiles ahead
    v16h a = load_fragA_f16(arow + kk, K, lane);
#pragma unroll
    for (int t = 0; t < 4; ++t) {
      v16h b = load_fragB_nk(Bt + (size_t)(n0 + 16 * t) * K + kk, K, lane);
      acc[t] = wmma16(a, b, acc[t]);
    }
  }
#pragma unroll
  for (int t = 0; t < 4; ++t) {
#pragma unroll
    for (int r = 0; r < 8; ++r) acc[t][r] *= scale;
    if constexpr (OUT_HALF)
      store_C_f16((_Float16*)Cv + (size_t)m0 * N + n0 + 16 * t, N, lane, acc[t]);
    else
      store_C_f32((float*)Cv     + (size_t)m0 * N + n0 + 16 * t, N, lane, acc[t]);
  }
}

// ---------------- kernel 3: fused attention + coverage ----------------
// grid = (LQ/16, B), 256 threads (8 wave32). Per block: 16 query rows, loop
// over all 8 heads. LDS: 16xLK score strip + 16xLK coverage accumulator +
// 16x64 P@V partial tile = 266.75 KB (<= 320 KB/WGP).
// After softmax the probabilities are re-packed f16 *in place* at the front of
// each score row, so the P@V A-fragments are raw ds_load_b128 (no converts).
__global__ void mha_attn_kernel(const _Float16* __restrict__ qproj,   // (B,LQ,D) f16, pre-scaled
                                const _Float16* __restrict__ kf,      // (B,LK,D) f16
                                const _Float16* __restrict__ vt,      // (B,H,DH,LK) f16
                                const unsigned char* __restrict__ mask, // (B,LQ,LK) bool
                                _Float16* __restrict__ attnout,       // (B,LQ,D) f16
                                float* __restrict__ coverage) {       // (B,LQ,LK) f32
  extern __shared__ float smem[];
  float* sS   = smem;                          // 16 x LDP scores -> f16 probs
  float* sCov = smem + (size_t)16 * LDP;       // 16 x LDP coverage accumulator
  float* sP   = smem + (size_t)32 * LDP;       // 16 x 64 P@V partial tile

  const int lane = threadIdx.x & 31;
  const int w    = threadIdx.x >> 5;
  const int b    = blockIdx.y;
  const int q0   = blockIdx.x * 16;

  for (int h = 0; h < H_; ++h) {
    // ---- S = Q_h @ K_h^T  (16 x LK), 8 waves x 16 column tiles ----
    const _Float16* qb = qproj + ((size_t)b * LQ_ + q0) * D_ + h * DH_;
    v16h qa0 = load_fragA_f16(qb,      D_, lane);
    v16h qa1 = load_fragA_f16(qb + 32, D_, lane);
    for (int kt = w; kt < LK_ / 16; kt += 8) {
      const _Float16* kb = kf + ((size_t)b * LK_ + kt * 16) * D_ + h * DH_;
      __builtin_prefetch(kb + (size_t)8 * 16 * D_, 0, 3);  // this wave's next tile
      v8f acc = {};
      acc = wmma16(qa0, load_fragB_nk(kb,      D_, lane), acc);
      acc = wmma16(qa1, load_fragB_nk(kb + 32, D_, lane), acc);
      store_C_f32(sS + kt * 16, LDP, lane, acc);
    }
    __syncthreads();

    // ---- masked softmax per row + coverage accumulation (wave owns 2 rows) --
    // 4-wide vectorized: lane owns 4 consecutive k per step (u32 mask word,
    // float4 LDS traffic).
#pragma unroll
    for (int rr = 0; rr < 2; ++rr) {
      const int r = w * 2 + rr;
      const int q = q0 + r;
      const unsigned char* mrow = mask + ((size_t)b * LQ_ + q) * LK_;
      float*     srow = sS + (size_t)r * LDP;
      _Float16*  prow = (_Float16*)srow;           // in-place f16 prob row
      float*     crow = sCov + (size_t)r * LDP;
      const float NEG = -__builtin_inff();

      float mx = NEG;
      for (int i = 0; i < LK_ / 128; ++i) {
        const int k0 = (i * 32 + lane) * 4;
        const unsigned int mw = *(const unsigned int*)(mrow + k0);
        float4 s = *(float4*)(srow + k0);
        if (mw & 0x000000ffu) s.x = NEG;
        if (mw & 0x0000ff00u) s.y = NEG;
        if (mw & 0x00ff0000u) s.z = NEG;
        if (mw & 0xff000000u) s.w = NEG;
        *(float4*)(srow + k0) = s;
        mx = fmaxf(mx, fmaxf(fmaxf(s.x, s.y), fmaxf(s.z, s.w)));
      }
      mx = wave_max32(mx);

      float l = 0.f;
      for (int i = 0; i < LK_ / 128; ++i) {
        const int k0 = (i * 32 + lane) * 4;
        float4 s = *(float4*)(srow + k0);
        float4 e;
        e.x = __expf(s.x - mx);  e.y = __expf(s.y - mx);   // exp(-inf)=0 masked
        e.z = __expf(s.z - mx);  e.w = __expf(s.w - mx);
        *(float4*)(srow + k0) = e;
        l += (e.x + e.y) + (e.z + e.w);
      }
      l = wave_sum32(l);
      const float inv = 1.f / l;

      for (int i = 0; i < LK_ / 128; ++i) {
        const int k0 = (i * 32 + lane) * 4;
        float4 e = *(float4*)(srow + k0);
        const float px = e.x * inv, py = e.y * inv, pz = e.z * inv, pw = e.w * inv;
        // coverage += p / H  (float4 RMW; this block owns these rows)
        float4 c;
        if (h == 0) {
          c.x = px * (1.f / H_); c.y = py * (1.f / H_);
          c.z = pz * (1.f / H_); c.w = pw * (1.f / H_);
        } else {
          c = *(float4*)(crow + k0);
          c.x += px * (1.f / H_); c.y += py * (1.f / H_);
          c.z += pz * (1.f / H_); c.w += pw * (1.f / H_);
        }
        *(float4*)(crow + k0) = c;
        // pack probs f16 in place (write offset 2k <= read offset 4k; wave LDS
        // ops are in-order, so the float4 read above already completed)
        v4h ph; ph[0] = (_Float16)px; ph[1] = (_Float16)py;
                ph[2] = (_Float16)pz; ph[3] = (_Float16)pw;
        *(v4h*)(prow + k0) = ph;
      }
    }
    __syncthreads();

    // ---- O_h = P @ V_h (16 x 64): all 8 waves; K split across wave pairs ----
    {
      const int dt   = w & 3;          // which 16-wide d tile
      const int half = w >> 2;         // which half of LK
      const _Float16* vtb = vt + ((size_t)(b * H_ + h) * DH_ + dt * 16) * LK_;
      const _Float16* pbase = (const _Float16*)sS;   // f16 rows, stride 2*LDP
      v8f acc = {};
      const int k0 = half * (LK_ / 2);
      for (int kk = k0; kk < k0 + LK_ / 2; kk += 32) {
        v16h a  = load_fragA_f16(pbase + kk, 2 * LDP, lane);  // ds_load_b128, no cvt
        v16h bb = load_fragB_nk(vtb + kk, LK_, lane);         // contiguous global
        acc = wmma16(a, bb, acc);
      }
      if (half == 1) store_C_f32(sP + dt * 16, 64, lane, acc);
      __syncthreads();
      if (half == 0) {
        const int n = lane & 15, mb = (lane >> 4) * 8;
#pragma unroll
        for (int r = 0; r < 8; ++r)
          acc[r] += sP[(size_t)(mb + r) * 64 + dt * 16 + n];
        store_C_f16(attnout + ((size_t)b * LQ_ + q0) * D_ + h * DH_ + dt * 16,
                    D_, lane, acc);
      }
      __syncthreads();   // also guards sS reuse for next head
    }
  }

  // ---- stream coverage tile to HBM (float4, written exactly once) ----
  float* covg = coverage + ((size_t)b * LQ_ + q0) * LK_;
  for (int idx = threadIdx.x; idx < 16 * (LK_ / 4); idx += 256) {
    const int r  = idx >> 9;              // / (LK_/4)
    const int k4 = (idx & (LK_ / 4 - 1)) * 4;
    *(float4*)(covg + (size_t)r * LK_ + k4) = *(float4*)(sCov + (size_t)r * LDP + k4);
  }
}

// ---------------- host-side launcher ----------------
extern "C" void kernel_launch(void* const* d_in, const int* in_sizes, int n_in,
                              void* d_out, int out_size, void* d_ws, size_t ws_size,
                              hipStream_t stream) {
  (void)in_sizes; (void)n_in; (void)out_size; (void)ws_size;

  const float* query = (const float*)d_in[0];
  const float* key   = (const float*)d_in[1];
  const float* value = (const float*)d_in[2];
  const unsigned char* mask = (const unsigned char*)d_in[3];  // jax bool -> 1 byte
  const float* Wq = (const float*)d_in[4];
  const float* Wc = (const float*)d_in[5];

  float* out      = (float*)d_out;                       // (B,LQ,D) f32
  float* coverage = out + (size_t)B_ * LQ_ * D_;         // (B,LQ,LK) f32

  const size_t NT = (size_t)B_ * LQ_ * D_;   // 4,194,304
  const size_t NW = (size_t)D_ * D_;         //   262,144

  _Float16* ws   = (_Float16*)d_ws;          // total ~41 MB of f16 scratch
  _Float16* qf   = ws;                       // query f16
  _Float16* kf   = qf   + NT;                // key f16
  _Float16* vt   = kf   + NT;                // value f16, transposed (B,H,DH,LK)
  _Float16* wqf  = vt   + NT;                // Wq f16
  _Float16* wcf  = wqf  + NW;                // Wc f16
  _Float16* qprj = wcf  + NW;                // scaled q projection f16
  _Float16* aout = qprj + NT;                // attention output f16

  // 1) convert inputs to f16 (V fused with head-transpose)
  mha_cvt_f16_kernel<<<dim3((NT + 255) / 256), 256, 0, stream>>>(query, qf, (int)NT);
  mha_cvt_f16_kernel<<<dim3((NT + 255) / 256), 256, 0, stream>>>(key,   kf, (int)NT);
  mha_cvt_transpose_v_kernel<<<dim3((NT + 255) / 256), 256, 0, stream>>>(value, vt);
  mha_cvt_f16_kernel<<<dim3((NW + 255) / 256), 256, 0, stream>>>(Wq, wqf, (int)NW);
  mha_cvt_f16_kernel<<<dim3((NW + 255) / 256), 256, 0, stream>>>(Wc, wcf, (int)NW);

  // 2) q = (query @ Wq^T) * DH^-0.5   (M=8192, N=K=512)
  dim3 gp(B_ * LQ_ / 16, D_ / (64 * 8));
  mha_gemm_nt_kernel<true><<<gp, 256, 0, stream>>>(qf, wqf, qprj,
                                                   B_ * LQ_, D_, D_, 0.125f);

  // 3) fused attention + softmax + coverage
  dim3 ga(LQ_ / 16, B_);
  const size_t smem_bytes = ((size_t)32 * LDP + 16 * 64) * sizeof(float);  // 266.75 KB
  mha_attn_kernel<<<ga, 256, smem_bytes, stream>>>(qprj, kf, vt, mask, aout, coverage);

  // 4) out = attn_out @ Wc^T  -> f32
  mha_gemm_nt_kernel<false><<<gp, 256, 0, stream>>>(aout, wcf, (void*)out,
                                                    B_ * LQ_, D_, D_, 1.0f);
}
